// CausalSelfAttention_68831145886167
// MI455X (gfx1250) — matrix operand order repaired and checked by reading, not compiled, not run on hardware
//
#include <hip/hip_runtime.h>
#include <cstdint>
#include <cstddef>

#define B_  2
#define T_  2048
#define C_  1024
#define H_  16
#define DH_ 64
#define BT_ (B_*T_)

typedef _Float16 v16h  __attribute__((ext_vector_type(16)));
typedef _Float16 half8 __attribute__((ext_vector_type(8)));
typedef float    v8f   __attribute__((ext_vector_type(8)));

__device__ __forceinline__ v8f v8f_zero() {
  v8f r;
#pragma unroll
  for (int i = 0; i < 8; ++i) r[i] = 0.0f;
  return r;
}

__device__ __forceinline__ v16h frag_cat(half8 lo, half8 hi) {
  v16h r;
#pragma unroll
  for (int i = 0; i < 8; ++i) { r[i] = lo[i]; r[i + 8] = hi[i]; }
  return r;
}

// A-operand fragment (16x32 f16). Lane holds row M = lane%16.
// lanes 0-15: K = kk+[0..7], kk+[16..23]; lanes 16-31: +8 / +24.
__device__ __forceinline__ v16h load_frag_a(const _Float16* base, int stride, int lane, int kk) {
  int row = lane & 15;
  int k0  = kk + ((lane & 16) ? 8 : 0);
  const _Float16* p = base + row * stride + k0;
  half8 lo = *(const half8*)(p);
  half8 hi = *(const half8*)(p + 16);
  return frag_cat(lo, hi);
}

// B-operand fragment (32x16 f16), sourced from an [N][K] row-major tile.
// Lane holds column N = lane%16; lanes 0-15: K = kk+[0..7], kk+[16..23]; lanes 16-31: +8 / +24.
__device__ __forceinline__ v16h load_frag_b(const _Float16* base, int stride, int lane, int kk) {
  int row = lane & 15;
  int k0  = kk + ((lane & 16) ? 8 : 0);
  const _Float16* p = base + row * stride + k0;
  half8 lo = *(const half8*)(p);
  half8 hi = *(const half8*)(p + 16);
  return frag_cat(lo, hi);
}

__device__ __forceinline__ v8f wmma16(v16h a, v16h b, v8f c) {
  return __builtin_amdgcn_wmma_f32_16x16x32_f16(false, a, false, b, (short)0, c, false, false);
}

// ---------------------------------------------------------------------------
// Fuse head-mix into QKV weights, convert all weights to f16.
// qw[h*DH+d][c] = sum_m mix[h][m] * w[(m*DH+d)*C + c]
// ---------------------------------------------------------------------------
__global__ __launch_bounds__(256) void k_prep_weights(
    const float* __restrict__ qw, const float* __restrict__ kw,
    const float* __restrict__ vw, const float* __restrict__ pw,
    const float* __restrict__ qmix, const float* __restrict__ kmix,
    const float* __restrict__ vmix,
    _Float16* __restrict__ qwh, _Float16* __restrict__ kwh,
    _Float16* __restrict__ vwh, _Float16* __restrict__ pwh) {
  const int CC = C_ * C_;
  int idx = blockIdx.x * blockDim.x + threadIdx.x;
  if (idx < 3 * CC) {
    int sel = idx / CC;
    int o   = idx % CC;
    int row = o / C_;
    int c   = o % C_;
    int h   = row / DH_;
    int d   = row % DH_;
    const float* w   = (sel == 0) ? qw : (sel == 1) ? kw : vw;
    const float* mix = (sel == 0) ? qmix : (sel == 1) ? kmix : vmix;
    float acc = 0.0f;
#pragma unroll
    for (int m = 0; m < H_; ++m) acc += mix[h * H_ + m] * w[(size_t)(m * DH_ + d) * C_ + c];
    _Float16* outp = (sel == 0) ? qwh : (sel == 1) ? kwh : vwh;
    outp[o] = (_Float16)acc;
  } else if (idx < 4 * CC) {
    int o = idx - 3 * CC;
    pwh[o] = (_Float16)pw[o];
  }
}

__global__ __launch_bounds__(256) void k_convert_x(const float* __restrict__ x,
                                                   _Float16* __restrict__ xh, int n) {
  int i = (blockIdx.x * blockDim.x + threadIdx.x) * 4;
  if (i + 3 < n) {
#pragma unroll
    for (int j = 0; j < 4; ++j) xh[i + j] = (_Float16)x[i + j];
  } else {
    for (int j = 0; j < 4 && i + j < n; ++j) xh[i + j] = (_Float16)x[i + j];
  }
}

// ---------------------------------------------------------------------------
// Per (b,t,h) gates: gv = 2*sigmoid(x[:32].w_ve), ga = sigmoid(x[:12].w_a)
// ---------------------------------------------------------------------------
__global__ __launch_bounds__(256) void k_gates(const float* __restrict__ x,
                                               const float* __restrict__ vegw,
                                               const float* __restrict__ agw,
                                               float* __restrict__ gv,
                                               float* __restrict__ ga) {
  int idx = blockIdx.x * blockDim.x + threadIdx.x;
  if (idx >= B_ * T_ * H_) return;
  int h  = idx % H_;
  int bt = idx / H_;
  const float* xr = x + (size_t)bt * C_;
  float a = 0.0f;
#pragma unroll
  for (int i = 0; i < 32; ++i) a += xr[i] * vegw[h * 32 + i];
  float g = 0.0f;
#pragma unroll
  for (int i = 0; i < 12; ++i) g += xr[i] * agw[h * 12 + i];
  gv[idx] = 2.0f / (1.0f + __expf(-a));
  ga[idx] = 1.0f / (1.0f + __expf(-g));
}

// ---------------------------------------------------------------------------
// Tiled WMMA GEMM: Cout[M,N] = A[M,K] * Bw[N,K]^T   (f16 in, OT out)
// Block tile 128x128, BK=64, 8 waves (2x4), each wave 64x32 -> 8 accumulators.
// Global->LDS is software-pipelined through registers: tile k+1 is fetched
// while WMMAs consume tile k from LDS.
// ---------------------------------------------------------------------------
template <typename OT>
__global__ __launch_bounds__(256) void k_gemm_f16(const _Float16* __restrict__ A,
                                                  const _Float16* __restrict__ Bw,
                                                  OT* __restrict__ Cout,
                                                  int M, int N, int K) {
  __shared__ __align__(16) _Float16 As[128][72];
  __shared__ __align__(16) _Float16 Bs[128][72];
  const int tid  = threadIdx.x;
  const int lane = tid & 31;
  const int wave = tid >> 5;
  const int wm   = wave >> 2;  // 0..1
  const int wn   = wave & 3;   // 0..3
  const int bm   = blockIdx.x;
  const int bn   = blockIdx.y;

  v8f acc[4][2];
#pragma unroll
  for (int i = 0; i < 4; ++i)
#pragma unroll
    for (int j = 0; j < 2; ++j) acc[i][j] = v8f_zero();

  half8 rA[4], rB[4];
  auto fetch = [&](int k0) {
#pragma unroll
    for (int it = 0; it < 4; ++it) {
      int e   = tid + it * 256;
      int row = e >> 3;
      int col = (e & 7) * 8;
      rA[it] = *(const half8*)(A  + (size_t)(bm * 128 + row) * K + k0 + col);
      rB[it] = *(const half8*)(Bw + (size_t)(bn * 128 + row) * K + k0 + col);
    }
  };
  fetch(0);

  for (int k0 = 0; k0 < K; k0 += 64) {
    __syncthreads();  // previous tile's readers done; LDS reusable
#pragma unroll
    for (int it = 0; it < 4; ++it) {
      int e   = tid + it * 256;
      int row = e >> 3;
      int col = (e & 7) * 8;
      *(half8*)&As[row][col] = rA[it];
      *(half8*)&Bs[row][col] = rB[it];
    }
    if (k0 + 64 < K) fetch(k0 + 64);  // overlap next-tile global loads with WMMAs
    __syncthreads();
#pragma unroll
    for (int kk = 0; kk < 64; kk += 32) {
      v16h af[4];
#pragma unroll
      for (int i = 0; i < 4; ++i) af[i] = load_frag_a(&As[wm * 64 + i * 16][0], 72, lane, kk);
      v16h bf[2];
#pragma unroll
      for (int j = 0; j < 2; ++j) bf[j] = load_frag_b(&Bs[wn * 32 + j * 16][0], 72, lane, kk);
#pragma unroll
      for (int i = 0; i < 4; ++i)
#pragma unroll
        for (int j = 0; j < 2; ++j) acc[i][j] = wmma16(af[i], bf[j], acc[i][j]);
    }
  }

  const int ln = lane & 15;
  const int hi = (lane >> 4) & 1;
#pragma unroll
  for (int i = 0; i < 4; ++i)
#pragma unroll
    for (int j = 0; j < 2; ++j)
#pragma unroll
      for (int r = 0; r < 8; ++r) {
        int row = bm * 128 + wm * 64 + i * 16 + r + 8 * hi;
        int col = bn * 128 + wn * 32 + j * 16 + ln;
        Cout[(size_t)row * N + col] = (OT)acc[i][j][r];
      }
}

// ---------------------------------------------------------------------------
// QKV epilogue per (b,t,h): v += gate_v*ve ; q,k: RoPE + RMS-norm;
// q additionally pre-scaled by 1/sqrt(DH) so flash uses raw WMMA scores;
// k second half shifted from token t-1 ("stationary") via recompute.
// ---------------------------------------------------------------------------
__global__ __launch_bounds__(256) void k_qkv_epi(const _Float16* __restrict__ qr,
                                                 const _Float16* __restrict__ kr,
                                                 const _Float16* __restrict__ vr,
                                                 const float* __restrict__ ve,
                                                 const float* __restrict__ cosp,
                                                 const float* __restrict__ sinp,
                                                 const float* __restrict__ gvArr,
                                                 _Float16* __restrict__ qh,
                                                 _Float16* __restrict__ kh,
                                                 _Float16* __restrict__ vh) {
  int idx = blockIdx.x * blockDim.x + threadIdx.x;
  if (idx >= B_ * T_ * H_) return;
  int h = idx % H_;
  int t = (idx / H_) % T_;
  int b = idx / (T_ * H_);
  size_t base = (size_t)idx * DH_;

  float gv = gvArr[idx];
  size_t vebase = (size_t)(b * T_ + t) * C_ + h * DH_;
#pragma unroll
  for (int d = 0; d < DH_; ++d)
    vh[base + d] = (_Float16)((float)vr[base + d] + gv * ve[vebase + d]);

  float tmp[DH_];
  // Q: RoPE + RMS (+ fold in attention scale 1/8)
  {
    const float* cp = cosp + t * 32;
    const float* sp = sinp + t * 32;
#pragma unroll
    for (int d = 0; d < 32; ++d) {
      float x1 = qr[base + d], x2 = qr[base + d + 32];
      tmp[d]      =  x1 * cp[d] + x2 * sp[d];
      tmp[d + 32] = -x1 * sp[d] + x2 * cp[d];
    }
    float ss = 0.0f;
#pragma unroll
    for (int d = 0; d < DH_; ++d) ss += tmp[d] * tmp[d];
    float rn = rsqrtf(ss * (1.0f / DH_) + 1e-6f) * 0.125f;  // * 1/sqrt(64)
#pragma unroll
    for (int d = 0; d < DH_; ++d) qh[base + d] = (_Float16)(tmp[d] * rn);
  }
  // K at t: first half
  {
    const float* cp = cosp + t * 32;
    const float* sp = sinp + t * 32;
#pragma unroll
    for (int d = 0; d < 32; ++d) {
      float x1 = kr[base + d], x2 = kr[base + d + 32];
      tmp[d]      =  x1 * cp[d] + x2 * sp[d];
      tmp[d + 32] = -x1 * sp[d] + x2 * cp[d];
    }
    float ss = 0.0f;
#pragma unroll
    for (int d = 0; d < DH_; ++d) ss += tmp[d] * tmp[d];
    float rn = rsqrtf(ss * (1.0f / DH_) + 1e-6f);
#pragma unroll
    for (int d = 0; d < 32; ++d) kh[base + d] = (_Float16)(tmp[d] * rn);
  }
  // K at max(t-1,0): second half (stationary shift)
  {
    int tp = (t > 0) ? (t - 1) : 0;
    size_t basep = ((size_t)(b * T_ + tp) * H_ + h) * DH_;
    const float* cp = cosp + tp * 32;
    const float* sp = sinp + tp * 32;
#pragma unroll
    for (int d = 0; d < 32; ++d) {
      float x1 = kr[basep + d], x2 = kr[basep + d + 32];
      tmp[d]      =  x1 * cp[d] + x2 * sp[d];
      tmp[d + 32] = -x1 * sp[d] + x2 * cp[d];
    }
    float ss = 0.0f;
#pragma unroll
    for (int d = 0; d < DH_; ++d) ss += tmp[d] * tmp[d];
    float rn = rsqrtf(ss * (1.0f / DH_) + 1e-6f);
#pragma unroll
    for (int d = 32; d < DH_; ++d) kh[base + d] = (_Float16)(tmp[d] * rn);
  }
}

// ---------------------------------------------------------------------------
// Flash attention without online max. RMS-norm bounds |score| <= 11.32, so
// p = exp(s - 8) never overflows f16 (p <= ~27) and the f32 row-sum is exact
// enough. Row-sum rides the matrix pipe via a ones-column B fragment.
// K/V tiles are software-pipelined through registers; all B fragments are
// preloaded into distinct registers so WMMAs overlap LDS loads.
// Grid (T/64, B*H), 128 threads = 4 waves; each wave: 16 query rows x DH=64.
// ---------------------------------------------------------------------------
__global__ __launch_bounds__(128) void k_flash(const _Float16* __restrict__ qh,
                                               const _Float16* __restrict__ kh,
                                               const _Float16* __restrict__ vh,
                                               const float* __restrict__ gaArr,
                                               const int* __restrict__ winp,
                                               _Float16* __restrict__ yh) {
  __shared__ __align__(16) _Float16 Kt[64][72];        // [key][d]
  __shared__ __align__(16) _Float16 Vt[64][72];        // [d][key] (transposed)
  __shared__ __align__(16) _Float16 Pt[4][16][72];     // per-wave P staging

  const int qtile = blockIdx.x;
  const int bh    = blockIdx.y;
  const int b     = bh / H_;
  const int h     = bh % H_;
  const int tid   = threadIdx.x;
  const int wave  = tid >> 5;
  const int lane  = tid & 31;
  const int ln    = lane & 15;
  const int hi    = (lane >> 4) & 1;
  const int TS    = H_ * DH_;  // token stride in halfs

  const int  W       = winp[0];
  const bool use_win = (W >= 0) && (W < T_ - 1);

  const _Float16* qbase = qh + ((size_t)(b * T_ + qtile * 64 + wave * 16) * H_ + h) * DH_;
  v16h qf0 = load_frag_a(qbase, TS, lane, 0);
  v16h qf1 = load_frag_a(qbase, TS, lane, 32);

  // ones-column B fragment: B[k][0] = 1, other columns 0
  v16h onesB;
  {
    _Float16 ov = (ln == 0) ? (_Float16)1.0f : (_Float16)0.0f;
#pragma unroll
    for (int i = 0; i < 16; ++i) onesB[i] = ov;
  }

  v8f O[4];
#pragma unroll
  for (int i = 0; i < 4; ++i) O[i] = v8f_zero();
  v8f Lacc = v8f_zero();  // column 0 holds row-sums

  int kt0 = 0;
  if (use_win) {
    int a = qtile * 64 - W - 63;
    if (a > 0) kt0 = a >> 6;
  }

  half8 rK[4], rV[4];
  auto fetchKV = [&](int kt) {
    const _Float16* kb = kh + ((size_t)(b * T_ + kt * 64) * H_ + h) * DH_;
    const _Float16* vb = vh + ((size_t)(b * T_ + kt * 64) * H_ + h) * DH_;
#pragma unroll
    for (int it = 0; it < 4; ++it) {
      int e   = tid + it * 128;
      int row = e >> 3;
      int col = (e & 7) * 8;
      rK[it] = *(const half8*)(kb + (size_t)row * TS + col);
      rV[it] = *(const half8*)(vb + (size_t)row * TS + col);
    }
  };
  fetchKV(kt0);

  for (int kt = kt0; kt <= qtile; ++kt) {
    __syncthreads();  // previous tile's readers done
#pragma unroll
    for (int it = 0; it < 4; ++it) {
      int e   = tid + it * 128;
      int row = e >> 3;
      int col = (e & 7) * 8;
      *(half8*)&Kt[row][col] = rK[it];
#pragma unroll
      for (int j = 0; j < 8; ++j) Vt[col + j][row] = rV[it][j];
    }
    if (kt < qtile) fetchKV(kt + 1);  // overlap next-tile global loads
    __syncthreads();

    // preload all K fragments (distinct regs -> loads pipeline past WMMAs)
    v16h kb0[4], kb1[4];
#pragma unroll
    for (int nt = 0; nt < 4; ++nt) {
      kb0[nt] = load_frag_b(&Kt[nt * 16][0], 72, lane, 0);
      kb1[nt] = load_frag_b(&Kt[nt * 16][0], 72, lane, 32);
    }
    // S = (Q/8) K^T  (scale pre-folded into Q)
    v8f S[4];
#pragma unroll
    for (int nt = 0; nt < 4; ++nt) {
      v8f s = v8f_zero();
      s = wmma16(qf0, kb0[nt], s);
      s = wmma16(qf1, kb1[nt], s);
      S[nt] = s;
    }

    // p = exp(s - 8), masked to 0 (causal / window)
    const bool need_mask = (kt == qtile) || use_win;
#pragma unroll
    for (int nt = 0; nt < 4; ++nt)
#pragma unroll
      for (int r = 0; r < 8; ++r) {
        float p = __expf(S[nt][r] - 8.0f);
        if (need_mask) {
          int row = qtile * 64 + wave * 16 + r + 8 * hi;
          int col = kt * 64 + nt * 16 + ln;
          bool ok = (col <= row) && (!use_win || (row - col) <= W);
          if (!ok) p = 0.0f;
        }
        S[nt][r] = p;
      }

    // stage P (C-layout -> A-layout via wave-private LDS; same-wave DS is in-order)
#pragma unroll
    for (int nt = 0; nt < 4; ++nt)
#pragma unroll
      for (int r = 0; r < 8; ++r)
        Pt[wave][r + 8 * hi][nt * 16 + ln] = (_Float16)S[nt][r];

    v16h pa0 = load_frag_a(&Pt[wave][0][0], 72, lane, 0);
    v16h pa1 = load_frag_a(&Pt[wave][0][0], 72, lane, 32);
    v16h vb0[4], vb1[4];
#pragma unroll
    for (int nt = 0; nt < 4; ++nt) {
      vb0[nt] = load_frag_b(&Vt[nt * 16][0], 72, lane, 0);
      vb1[nt] = load_frag_b(&Vt[nt * 16][0], 72, lane, 32);
    }
#pragma unroll
    for (int nt = 0; nt < 4; ++nt) {
      O[nt] = wmma16(pa0, vb0[nt], O[nt]);
      O[nt] = wmma16(pa1, vb1[nt], O[nt]);
    }
    // row-sum on the matrix pipe: L += P @ ones_col
    Lacc = wmma16(pa0, onesB, Lacc);
    Lacc = wmma16(pa1, onesB, Lacc);
  }

  // epilogue: broadcast row-sums from column-0 lanes, apply gate, write f16 y
#pragma unroll
  for (int r = 0; r < 8; ++r) {
    float l   = __shfl(Lacc[r], hi ? 16 : 0, 32);
    int tok   = qtile * 64 + wave * 16 + r + 8 * hi;
    float ga  = gaArr[(size_t)(b * T_ + tok) * H_ + h];
    float inv = ga / l;
#pragma unroll
    for (int nt = 0; nt < 4; ++nt)
      yh[((size_t)(b * T_ + tok) * H_ + h) * DH_ + nt * 16 + ln] =
          (_Float16)(O[nt][r] * inv);
  }
}

// ---------------------------------------------------------------------------
extern "C" void kernel_launch(void* const* d_in, const int* in_sizes, int n_in,
                              void* d_out, int out_size, void* d_ws, size_t ws_size,
                              hipStream_t stream) {
  const float* x    = (const float*)d_in[0];
  const float* ve   = (const float*)d_in[1];
  const float* cosp = (const float*)d_in[2];
  const float* sinp = (const float*)d_in[3];
  const float* cqw  = (const float*)d_in[4];
  const float* ckw  = (const float*)d_in[5];
  const float* cvw  = (const float*)d_in[6];
  const float* cpw  = (const float*)d_in[7];
  const float* vegw = (const float*)d_in[8];
  const float* agw  = (const float*)d_in[9];
  const float* qmix = (const float*)d_in[10];
  const float* kmix = (const float*)d_in[11];
  const float* vmix = (const float*)d_in[12];
  const int*   win  = (const int*)d_in[13];
  float* out = (float*)d_out;

  const size_t BTC = (size_t)B_ * T_ * C_;
  const size_t CC  = (size_t)C_ * C_;
  const size_t BTH = (size_t)B_ * T_ * H_;

  uint8_t* w = (uint8_t*)d_ws;
  auto take = [&](size_t bytes) {
    uint8_t* p = w;
    w += (bytes + 255) & ~(size_t)255;
    return p;
  };
  _Float16* x_h  = (_Float16*)take(BTC * 2);
  _Float16* qw_h = (_Float16*)take(CC * 2);
  _Float16* kw_h = (_Float16*)take(CC * 2);
  _Float16* vw_h = (_Float16*)take(CC * 2);
  _Float16* pw_h = (_Float16*)take(CC * 2);
  _Float16* q_r  = (_Float16*)take(BTC * 2);
  _Float16* k_r  = (_Float16*)take(BTC * 2);
  _Float16* v_r  = (_Float16*)take(BTC * 2);
  _Float16* qh   = (_Float16*)take(BTC * 2);
  _Float16* kh   = (_Float16*)take(BTC * 2);
  _Float16* vh   = (_Float16*)take(BTC * 2);
  _Float16* yh   = (_Float16*)take(BTC * 2);
  float*    gvA  = (float*)take(BTH * 4);
  float*    gaA  = (float*)take(BTH * 4);

  k_prep_weights<<<(int)((4 * CC + 255) / 256), 256, 0, stream>>>(
      cqw, ckw, cvw, cpw, qmix, kmix, vmix, qw_h, kw_h, vw_h, pw_h);
  k_convert_x<<<(int)((BTC / 4 + 255) / 256), 256, 0, stream>>>(x, x_h, (int)BTC);
  k_gates<<<(int)((BTH + 255) / 256), 256, 0, stream>>>(x, vegw, agw, gvA, gaA);

  dim3 ggrid(BT_ / 128, C_ / 128);  // 32 x 8
  k_gemm_f16<_Float16><<<ggrid, 256, 0, stream>>>(x_h, qw_h, q_r, BT_, C_, C_);
  k_gemm_f16<_Float16><<<ggrid, 256, 0, stream>>>(x_h, kw_h, k_r, BT_, C_, C_);
  k_gemm_f16<_Float16><<<ggrid, 256, 0, stream>>>(x_h, vw_h, v_r, BT_, C_, C_);

  k_qkv_epi<<<(int)((BTH + 255) / 256), 256, 0, stream>>>(
      q_r, k_r, v_r, ve, cosp, sinp, gvA, qh, kh, vh);

  dim3 fgrid(T_ / 64, B_ * H_);  // 32 x 32
  k_flash<<<fgrid, 128, 0, stream>>>(qh, kh, vh, gaA, win, yh);

  k_gemm_f16<float><<<ggrid, 256, 0, stream>>>(yh, pw_h, out, BT_, C_, C_);
}
